// CWVAE_58213986730525
// MI455X (gfx1250) — compile-verified, matrix operand order checked by reading, not run on hardware
//
#include <hip/hip_runtime.h>
#include <hip/hip_bf16.h>

// ---------------- types ----------------
typedef __bf16 bf16_t;
typedef __attribute__((ext_vector_type(16))) __bf16 v16bf;
typedef __attribute__((ext_vector_type(8)))  __bf16 v8bf;
typedef __attribute__((ext_vector_type(8)))  float  v8f;

#define Bsz    128
#define STOCH  128
#define DETER  1024
#define EMBED  512
#define OBS_D  512

// ---------------- weight packing ----------------
// Packed layout per weight W[K,N] (row-major, stride N):
//   tile order [kc][nt] with kc over K/32, nt over N/16; each tile = 512 bf16:
//   element (lane, i) = W[kc*32 + kmap(lane,i)][nt*16 + (lane&15)]
//   kmap(lane,i) = (lane>>4)*8 + (i<8 ? i : i+8)   (CDNA5 16-bit frag map)
// The 4 consecutive nt tiles a wave consumes per k-chunk are contiguous (4KB).
__global__ void pack_weight(const float* __restrict__ W, bf16_t* __restrict__ Wpk,
                            int K, int N) {
    int total = K * N;
    int ntn = N >> 4;
    for (int idx = blockIdx.x * blockDim.x + threadIdx.x; idx < total;
         idx += gridDim.x * blockDim.x) {
        int e  = idx & 511;
        int t  = idx >> 9;
        int kc = t / ntn;
        int nt = t - kc * ntn;
        int lane = e >> 4, i = e & 15;
        int kk = (kc << 5) + ((lane >> 4) << 3) + (i < 8 ? i : i + 8);
        int nn = (nt << 4) + (lane & 15);
        Wpk[idx] = (bf16_t)W[(size_t)kk * N + nn];
    }
}

__global__ void f32_to_bf16_kernel(const float* __restrict__ src,
                                   bf16_t* __restrict__ dst, int n) {
    for (int i = blockIdx.x * blockDim.x + threadIdx.x; i < n;
         i += gridDim.x * blockDim.x)
        dst[i] = (bf16_t)src[i];
}

// ---------------- WMMA GEMM core ----------------
// C[128,N] = concat(A0[:, :K0], A1[:, :Ktot-K0]) @ Wpk + bias, opt. relu.
// One wave32 owns a 16(M) x 64(N) strip: 1 A-frag load feeds 4 WMMAs.
__device__ __forceinline__ v16bf load_a_frag(const bf16_t* __restrict__ src,
                                             int lda, int row, int col0) {
    const v8bf lo = *(const v8bf*)(src + (size_t)row * lda + col0);
    const v8bf hi = *(const v8bf*)(src + (size_t)row * lda + col0 + 16);
    v16bf r;
#pragma unroll
    for (int i = 0; i < 8; ++i) { r[i] = lo[i]; r[i + 8] = hi[i]; }
    return r;
}

__device__ __forceinline__ void gemm_core(
        const bf16_t* __restrict__ A0, int lda0, int K0,
        const bf16_t* __restrict__ A1, int lda1,
        const bf16_t* __restrict__ Wpk, const float* __restrict__ bias,
        float* __restrict__ outF, bf16_t* __restrict__ outBf,
        int N, int Ktot, int relu, int tile, int lane) {
    const int nstrips = N >> 6;              // strips of 4 n-tiles
    if (tile >= nstrips * 8) return;         // uniform per wave
    const int mt  = tile & 7;                // 128/16 row tiles
    const int nt0 = (tile >> 3) << 2;        // first of 4 n-tiles
    const int ntn = N >> 4;

    const int row  = (mt << 4) + (lane & 15);
    const int base = (lane >> 4) << 3;
    const bf16_t* wbase = Wpk + (size_t)nt0 * 512 + lane * 16;

    v8f c[4] = {};
    const int nk0 = K0 >> 5;
    const int nk1 = (Ktot - K0) >> 5;
    int kc = 0;
    for (int k0 = 0; k0 < nk0; ++k0, ++kc) {
        v16bf a = load_a_frag(A0, lda0, row, (k0 << 5) + base);
        const bf16_t* wp = wbase + (size_t)kc * ntn * 512;
#pragma unroll
        for (int j = 0; j < 4; ++j) {
            v16bf b = *(const v16bf*)(wp + (size_t)j * 512);
            c[j] = __builtin_amdgcn_wmma_f32_16x16x32_bf16(
                       false, a, false, b, (short)0, c[j], false, false);
        }
    }
    for (int k1 = 0; k1 < nk1; ++k1, ++kc) {
        v16bf a = load_a_frag(A1, lda1, row, (k1 << 5) + base);
        const bf16_t* wp = wbase + (size_t)kc * ntn * 512;
#pragma unroll
        for (int j = 0; j < 4; ++j) {
            v16bf b = *(const v16bf*)(wp + (size_t)j * 512);
            c[j] = __builtin_amdgcn_wmma_f32_16x16x32_bf16(
                       false, a, false, b, (short)0, c[j], false, false);
        }
    }

    const int row0 = (mt << 4) + ((lane >> 4) << 3);
#pragma unroll
    for (int j = 0; j < 4; ++j) {
        const int coln = ((nt0 + j) << 4) + (lane & 15);
        const float bv = bias ? bias[coln] : 0.0f;
#pragma unroll
        for (int v = 0; v < 8; ++v) {
            float x = c[j][v] + bv;
            if (relu) x = fmaxf(x, 0.0f);
            size_t idx = (size_t)(row0 + v) * N + coln;
            if (outF)  outF[idx]  = x;
            if (outBf) outBf[idx] = (bf16_t)x;
        }
    }
}

__global__ void gemm_wmma_bf16(const bf16_t* __restrict__ A0, int lda0, int K0,
                               const bf16_t* __restrict__ A1, int lda1,
                               const bf16_t* __restrict__ Wpk,
                               const float* __restrict__ bias,
                               float* __restrict__ outF,
                               bf16_t* __restrict__ outBf,
                               int N, int Ktot, int relu) {
    gemm_core(A0, lda0, K0, A1, lda1, Wpk, bias, outF, outBf, N, Ktot, relu,
              blockIdx.x * 4 + (threadIdx.x >> 5), threadIdx.x & 31);
}

// gi = h @ Wih + bih  (y==0)   |   gh = det @ Whh + bhh  (y==1), one launch
__global__ void gemm_gates(const bf16_t* __restrict__ hBf,
                           const bf16_t* __restrict__ detBf,
                           const bf16_t* __restrict__ WihPk,
                           const bf16_t* __restrict__ WhhPk,
                           const float* __restrict__ bih,
                           const float* __restrict__ bhh,
                           float* __restrict__ giF, float* __restrict__ ghF) {
    const int tile = blockIdx.x * 4 + (threadIdx.x >> 5);
    const int lane = threadIdx.x & 31;
    if (blockIdx.y == 0)
        gemm_core(hBf, EMBED, EMBED, nullptr, 0, WihPk, bih, giF, nullptr,
                  3 * DETER, EMBED, 0, tile, lane);
    else
        gemm_core(detBf, DETER, DETER, nullptr, 0, WhhPk, bhh, ghF, nullptr,
                  3 * DETER, DETER, 0, tile, lane);
}

// ---------------- GRU pointwise ----------------
__global__ void gru_pointwise(const float* __restrict__ gi,
                              const float* __restrict__ gh,
                              float* __restrict__ detF,
                              bf16_t* __restrict__ detBf,
                              bf16_t* __restrict__ seqBf,
                              float* __restrict__ seqF,
                              int T, int t) {
    int idx = blockIdx.x * blockDim.x + threadIdx.x;
    if (idx >= Bsz * DETER) return;
    int b = idx >> 10, j = idx & 1023;
    size_t g = (size_t)b * 3 * DETER;
    float ir = gi[g + j],             hr = gh[g + j];
    float iz = gi[g + DETER + j],     hz = gh[g + DETER + j];
    float in = gi[g + 2 * DETER + j], hn = gh[g + 2 * DETER + j];
    float r = 1.0f / (1.0f + __expf(-(ir + hr)));
    float z = 1.0f / (1.0f + __expf(-(iz + hz)));
    float n = tanhf(in + r * hn);
    float dn = (1.0f - z) * n + z * detF[idx];
    detF[idx]  = dn;
    detBf[idx] = (bf16_t)dn;
    size_t o = (((size_t)b * T + t) << 10) + j;
    if (seqBf) seqBf[o] = (bf16_t)dn;
    if (seqF)  seqF[o]  = dn;
}

// ---------------- host driver ----------------
extern "C" void kernel_launch(void* const* d_in, const int* in_sizes, int n_in,
                              void* d_out, int out_size, void* d_ws, size_t ws_size,
                              hipStream_t stream) {
    (void)in_sizes; (void)n_in; (void)out_size; (void)ws_size;
    const float* obs[3] = { (const float*)d_in[0], (const float*)d_in[1],
                            (const float*)d_in[2] };
    const float* Wp  = (const float*)d_in[3];
    const float* bp  = (const float*)d_in[4];
    const float* Wih = (const float*)d_in[5];
    const float* Whh = (const float*)d_in[6];
    const float* bih = (const float*)d_in[7];
    const float* bhh = (const float*)d_in[8];
    const float* Wq  = (const float*)d_in[9];
    const float* bq  = (const float*)d_in[10];
    const float* Wqm = (const float*)d_in[11];
    const float* bqm = (const float*)d_in[12];
    float* out = (float*)d_out;

    const int Tlev[3] = {256, 64, 16};

    // --- carve workspace ---
    char* p = (char*)d_ws;
    auto carve = [&](size_t bytes) -> void* {
        void* r = (void*)p;
        p += (bytes + 255) & ~(size_t)255;
        return r;
    };
    const size_t szWp  = (size_t)(STOCH + DETER) * EMBED;     // 1152*512
    const size_t szWih = (size_t)EMBED * 3 * DETER;           // 512*3072
    const size_t szWhh = (size_t)DETER * 3 * DETER;           // 1024*3072
    const size_t szWq  = (size_t)(DETER + OBS_D) * EMBED;     // 1536*512
    const size_t szWqm = (size_t)EMBED * STOCH;               // 512*128

    bf16_t* WpPk  = (bf16_t*)carve(3 * szWp  * 2);
    bf16_t* WihPk = (bf16_t*)carve(3 * szWih * 2);
    bf16_t* WhhPk = (bf16_t*)carve(3 * szWhh * 2);
    bf16_t* WqPk  = (bf16_t*)carve(3 * szWq  * 2);
    bf16_t* WqmPk = (bf16_t*)carve(3 * szWqm * 2);

    bf16_t* obsBf[3];
    for (int l = 0; l < 3; ++l)
        obsBf[l] = (bf16_t*)carve((size_t)Bsz * Tlev[l] * OBS_D * 2);

    bf16_t* dets1 = (bf16_t*)carve((size_t)Bsz * 64 * DETER * 2); // level-1 dets
    bf16_t* dets2 = (bf16_t*)carve((size_t)Bsz * 16 * DETER * 2); // level-2 dets
    bf16_t* zeroCtx = (bf16_t*)carve((size_t)Bsz * DETER * 2);

    bf16_t* sampleBf = (bf16_t*)carve((size_t)Bsz * STOCH * 2);
    bf16_t* detBf    = (bf16_t*)carve((size_t)Bsz * DETER * 2);
    float*  detF     = (float*) carve((size_t)Bsz * DETER * 4);
    bf16_t* hBf      = (bf16_t*)carve((size_t)Bsz * EMBED * 2);
    bf16_t* qhBf     = (bf16_t*)carve((size_t)Bsz * EMBED * 2);
    float*  giF      = (float*) carve((size_t)Bsz * 3 * DETER * 4);
    float*  ghF      = (float*) carve((size_t)Bsz * 3 * DETER * 4);

    // --- one-time packing / conversion ---
    for (int l = 0; l < 3; ++l) {
        pack_weight<<<512, 256, 0, stream>>>(Wp  + l * szWp,  WpPk  + l * szWp,
                                             STOCH + DETER, EMBED);
        pack_weight<<<512, 256, 0, stream>>>(Wih + l * szWih, WihPk + l * szWih,
                                             EMBED, 3 * DETER);
        pack_weight<<<512, 256, 0, stream>>>(Whh + l * szWhh, WhhPk + l * szWhh,
                                             DETER, 3 * DETER);
        pack_weight<<<512, 256, 0, stream>>>(Wq  + l * szWq,  WqPk  + l * szWq,
                                             DETER + OBS_D, EMBED);
        pack_weight<<<128, 256, 0, stream>>>(Wqm + l * szWqm, WqmPk + l * szWqm,
                                             EMBED, STOCH);
        int n = Bsz * Tlev[l] * OBS_D;
        f32_to_bf16_kernel<<<1024, 256, 0, stream>>>(obs[l], obsBf[l], n);
    }
    hipMemsetAsync(zeroCtx, 0, (size_t)Bsz * DETER * 2, stream);

    // --- levels, coarse -> fine ---
    for (int level = 2; level >= 0; --level) {
        const int T = Tlev[level];
        hipMemsetAsync(sampleBf, 0, (size_t)Bsz * STOCH * 2, stream);
        hipMemsetAsync(detBf,    0, (size_t)Bsz * DETER * 2, stream);
        hipMemsetAsync(detF,     0, (size_t)Bsz * DETER * 4, stream);

        const bf16_t* ctxBuf; int ctxStride, Tprev;
        if (level == 2)      { ctxBuf = zeroCtx; ctxStride = DETER;      Tprev = 1;  }
        else if (level == 1) { ctxBuf = dets2;   ctxStride = 16 * DETER; Tprev = 16; }
        else                 { ctxBuf = dets1;   ctxStride = 64 * DETER; Tprev = 64; }

        const bf16_t* WpL  = WpPk  + level * szWp;
        const bf16_t* WihL = WihPk + level * szWih;
        const bf16_t* WhhL = WhhPk + level * szWhh;
        const bf16_t* WqL  = WqPk  + level * szWq;
        const bf16_t* WqmL = WqmPk + level * szWqm;
        const float* bpL  = bp  + level * EMBED;
        const float* bihL = bih + level * 3 * DETER;
        const float* bhhL = bhh + level * 3 * DETER;
        const float* bqL  = bq  + level * EMBED;
        const float* bqmL = bqm + level * STOCH;

        bf16_t* seqBf = (level == 1) ? dets1 : (level == 2) ? dets2 : nullptr;
        float*  seqF  = (level == 0) ? out : nullptr;

        for (int t = 0; t < T; ++t) {
            // h = relu([sample | ctx_t] @ Wp + bp)        -> hBf [128,512]
            gemm_wmma_bf16<<<16, 128, 0, stream>>>(
                sampleBf, STOCH, STOCH,
                ctxBuf + (size_t)(t % Tprev) * DETER, ctxStride,
                WpL, bpL, nullptr, hBf, EMBED, STOCH + DETER, 1);
            // gi = h @ Wih + bih ; gh = det @ Whh + bhh   -> giF/ghF [128,3072]
            gemm_gates<<<dim3(96, 2), 128, 0, stream>>>(
                hBf, detBf, WihL, WhhL, bihL, bhhL, giF, ghF);
            // GRU gates -> det (f32 + bf16), record det sequence
            gru_pointwise<<<512, 256, 0, stream>>>(giF, ghF, detF, detBf,
                                                   seqBf, seqF, T, t);
            // qh = relu([det | obs_t] @ Wq + bq)          -> qhBf [128,512]
            gemm_wmma_bf16<<<16, 128, 0, stream>>>(
                detBf, DETER, DETER,
                obsBf[level] + (size_t)t * OBS_D, T * OBS_D,
                WqL, bqL, nullptr, qhBf, EMBED, DETER + OBS_D, 1);
            // sample = qh @ Wqm + bqm                     -> sampleBf [128,128]
            gemm_wmma_bf16<<<4, 128, 0, stream>>>(
                qhBf, EMBED, EMBED, nullptr, 0,
                WqmL, bqmL, nullptr, sampleBf, STOCH, EMBED, 0);
        }
    }
}
